// Weight_2765958938636
// MI455X (gfx1250) — matrix operand order repaired and checked by reading, not compiled, not run on hardware
//
#include <hip/hip_runtime.h>
#include <hip/hip_bf16.h>
#include <math.h>

// ---------------------------------------------------------------------------
// Types for CDNA5 WMMA (wave32): v_wmma_f32_16x16x32_bf16
// ---------------------------------------------------------------------------
typedef __attribute__((ext_vector_type(16))) __bf16 v16bf;
typedef __attribute__((ext_vector_type(8)))  __bf16 v8bf;
typedef __attribute__((ext_vector_type(8)))  float  v8f;

#define LDS_STRIDE 48          // elements (96 B): keeps fragment chunks 16B-aligned
#define LDS_BUF (128 * LDS_STRIDE)

__device__ __forceinline__ unsigned short f32_to_bf16_rte(float f) {
    unsigned u = __builtin_bit_cast(unsigned, f);
    unsigned r = 0x7FFFu + ((u >> 16) & 1u);
    u += r;
    return (unsigned short)(u >> 16);
}
__device__ __forceinline__ float bf16u_to_f32(unsigned short h) {
    unsigned u = ((unsigned)h) << 16;
    return __builtin_bit_cast(float, u);
}
__device__ __forceinline__ unsigned pk2(float a, float b) {
    return (unsigned)f32_to_bf16_rte(a) | ((unsigned)f32_to_bf16_rte(b) << 16);
}

// A fragment: lane l holds row (l&15); K = (l>>4)*8 + {0..7}, then +16.
__device__ __forceinline__ v16bf load_a_frag(const unsigned short* row, int lhi) {
    const unsigned short* p = row + lhi * 8;
    v8bf a0 = *reinterpret_cast<const v8bf*>(p);
    v8bf a1 = *reinterpret_cast<const v8bf*>(p + 16);
    return __builtin_shufflevector(a0, a1, 0,1,2,3,4,5,6,7,8,9,10,11,12,13,14,15);
}
// B fragment from N-major (transposed) LDS tile: lane l holds col (l&15);
// K = (l>>4)*8 + {0..7}, then +16.
__device__ __forceinline__ v16bf load_b_frag(const unsigned short* row, int lhi) {
    const unsigned short* p = row + lhi * 8;
    v8bf b0 = *reinterpret_cast<const v8bf*>(p);
    v8bf b1 = *reinterpret_cast<const v8bf*>(p + 16);
    return __builtin_shufflevector(b0, b1, 0,1,2,3,4,5,6,7,8,9,10,11,12,13,14,15);
}

// ---------------------------------------------------------------------------
// Row L2 norms: one wave per row of a [nrows x ncols] f32 matrix.
// ---------------------------------------------------------------------------
__global__ __launch_bounds__(256) void rownorm_kernel(const float* __restrict__ x,
                                                      float* __restrict__ nrm,
                                                      int nrows, int ncols) {
    int wid  = threadIdx.x >> 5;
    int lane = threadIdx.x & 31;
    int row  = blockIdx.x * 8 + wid;
    if (row >= nrows) return;
    const float* p = x + (size_t)row * ncols;
    float s = 0.f;
    for (int c = lane; c < ncols; c += 32) { float v = p[c]; s += v * v; }
    #pragma unroll
    for (int off = 16; off > 0; off >>= 1) s += __shfl_xor(s, off, 32);
    if (lane == 0) nrm[row] = sqrtf(s);
}

__global__ __launch_bounds__(256) void cvt_bf16_kernel(const float* __restrict__ in,
                                                       unsigned short* __restrict__ out,
                                                       int n) {
    int i = blockIdx.x * 256 + threadIdx.x;
    if (i < n) out[i] = f32_to_bf16_rte(in[i]);
}

// ---------------------------------------------------------------------------
// GEMM 1: S[b,q,k] = (query[b,q,:]·key[b,k,:]) / (Qn*Kn + 1e-2), bf16 out.
// (1/100 input scaling folds exactly into epsilon: 1e-6*1e4 = 1e-2.)
// 256 thr (8 waves), 128x128 tile, wave = 64x32 = 4x2 WMMA accums.
// Software-pipelined double-buffered LDS: one barrier per K-step.
// ---------------------------------------------------------------------------
__global__ __launch_bounds__(256)
void gemm_qk_kernel(const float* __restrict__ Q, const float* __restrict__ Kx,
                    const float* __restrict__ Qn, const float* __restrict__ Kn,
                    unsigned short* __restrict__ S) {
    const int M = 1024, N = 1024, Kd = 512, NT = Kd / 32;
    __shared__ __align__(16) unsigned short sA[2 * LDS_BUF];
    __shared__ __align__(16) unsigned short sB[2 * LDS_BUF];

    const int b = blockIdx.z;
    const int m0 = blockIdx.y * 128, n0 = blockIdx.x * 128;
    const float* Ab = Q  + (size_t)b * M * Kd;
    const float* Bb = Kx + (size_t)b * N * Kd;

    const int tid  = threadIdx.x;
    const int lane = tid & 31, w = tid >> 5;
    const int lm = lane & 15, lhi = lane >> 4;
    const int waveM = (w >> 2) * 64;  // 0,64
    const int waveN = (w & 3) * 32;   // 0,32,64,96
    const int lrow = tid >> 1, lhalf = tid & 1;

    const float* ga = Ab + (size_t)(m0 + lrow) * Kd + lhalf * 16;
    const float* gb = Bb + (size_t)(n0 + lrow) * Kd + lhalf * 16;
    unsigned short* const la = sA + lrow * LDS_STRIDE + lhalf * 16;
    unsigned short* const lb = sB + lrow * LDS_STRIDE + lhalf * 16;

    v8f acc[4][2] = {};
    float4 ra0, ra1, ra2, ra3, rb0, rb1, rb2, rb3;

    // prefetch tile 0
    {
        const float4* a4 = reinterpret_cast<const float4*>(ga);
        const float4* b4 = reinterpret_cast<const float4*>(gb);
        ra0 = a4[0]; ra1 = a4[1]; ra2 = a4[2]; ra3 = a4[3];
        rb0 = b4[0]; rb1 = b4[1]; rb2 = b4[2]; rb3 = b4[3];
    }
    // spill tile 0 -> buffer 0
    {
        uint4 oa0 = { pk2(ra0.x,ra0.y), pk2(ra0.z,ra0.w), pk2(ra1.x,ra1.y), pk2(ra1.z,ra1.w) };
        uint4 oa1 = { pk2(ra2.x,ra2.y), pk2(ra2.z,ra2.w), pk2(ra3.x,ra3.y), pk2(ra3.z,ra3.w) };
        uint4 ob0 = { pk2(rb0.x,rb0.y), pk2(rb0.z,rb0.w), pk2(rb1.x,rb1.y), pk2(rb1.z,rb1.w) };
        uint4 ob1 = { pk2(rb2.x,rb2.y), pk2(rb2.z,rb2.w), pk2(rb3.x,rb3.y), pk2(rb3.z,rb3.w) };
        reinterpret_cast<uint4*>(la)[0] = oa0; reinterpret_cast<uint4*>(la)[1] = oa1;
        reinterpret_cast<uint4*>(lb)[0] = ob0; reinterpret_cast<uint4*>(lb)[1] = ob1;
    }

    int cur = 0;
    for (int t = 0; t < NT; ++t) {
        __syncthreads();
        if (t + 1 < NT) {  // issue global loads for next tile early
            const float4* a4 = reinterpret_cast<const float4*>(ga + (t + 1) * 32);
            const float4* b4 = reinterpret_cast<const float4*>(gb + (t + 1) * 32);
            ra0 = a4[0]; ra1 = a4[1]; ra2 = a4[2]; ra3 = a4[3];
            rb0 = b4[0]; rb1 = b4[1]; rb2 = b4[2]; rb3 = b4[3];
        }
        const unsigned short* cA = sA + cur * LDS_BUF;
        const unsigned short* cB = sB + cur * LDS_BUF;
        v16bf bf0 = load_b_frag(cB + (waveN + 0  + lm) * LDS_STRIDE, lhi);
        v16bf bf1 = load_b_frag(cB + (waveN + 16 + lm) * LDS_STRIDE, lhi);
        #pragma unroll
        for (int sm = 0; sm < 4; sm++) {
            v16bf af = load_a_frag(cA + (waveM + sm * 16 + lm) * LDS_STRIDE, lhi);
            acc[sm][0] = __builtin_amdgcn_wmma_f32_16x16x32_bf16(
                false, af, false, bf0, (short)0, acc[sm][0], false, false);
            acc[sm][1] = __builtin_amdgcn_wmma_f32_16x16x32_bf16(
                false, af, false, bf1, (short)0, acc[sm][1], false, false);
        }
        if (t + 1 < NT) {  // spill next tile into the other buffer
            unsigned short* na = la + (cur ^ 1) * LDS_BUF;
            unsigned short* nb = lb + (cur ^ 1) * LDS_BUF;
            uint4 oa0 = { pk2(ra0.x,ra0.y), pk2(ra0.z,ra0.w), pk2(ra1.x,ra1.y), pk2(ra1.z,ra1.w) };
            uint4 oa1 = { pk2(ra2.x,ra2.y), pk2(ra2.z,ra2.w), pk2(ra3.x,ra3.y), pk2(ra3.z,ra3.w) };
            uint4 ob0 = { pk2(rb0.x,rb0.y), pk2(rb0.z,rb0.w), pk2(rb1.x,rb1.y), pk2(rb1.z,rb1.w) };
            uint4 ob1 = { pk2(rb2.x,rb2.y), pk2(rb2.z,rb2.w), pk2(rb3.x,rb3.y), pk2(rb3.z,rb3.w) };
            reinterpret_cast<uint4*>(na)[0] = oa0; reinterpret_cast<uint4*>(na)[1] = oa1;
            reinterpret_cast<uint4*>(nb)[0] = ob0; reinterpret_cast<uint4*>(nb)[1] = ob1;
        }
        cur ^= 1;
    }

    #pragma unroll
    for (int sm = 0; sm < 4; sm++)
        #pragma unroll
        for (int sn = 0; sn < 2; sn++) {
            int gn = n0 + waveN + sn * 16 + lm;
            float kn = Kn[b * N + gn];
            #pragma unroll
            for (int r = 0; r < 8; r++) {
                int gm = m0 + waveM + sm * 16 + lhi * 8 + r;
                float qn = Qn[b * M + gm];
                float v  = acc[sm][sn][r] * __builtin_amdgcn_rcpf(qn * kn + 1e-2f);
                S[((size_t)b * M + gm) * N + gn] = f32_to_bf16_rte(v);
            }
        }
}

// ---------------------------------------------------------------------------
// GEMM 2/3: D[b,q,o] = act( sum_k A[b,q,k] * W[o,k] ), all bf16, W shared.
// W stored [out,in] row-major == already the N-major tile layout we want.
// Same double-buffered pipeline as above.
// ---------------------------------------------------------------------------
__global__ __launch_bounds__(256)
void gemm_mlp_kernel(const unsigned short* __restrict__ Ain,
                     const unsigned short* __restrict__ Wt,
                     unsigned short* __restrict__ Dout, int applyGelu) {
    const int M = 1024, N = 1024, Kd = 1024, NT = Kd / 32;
    __shared__ __align__(16) unsigned short sA[2 * LDS_BUF];
    __shared__ __align__(16) unsigned short sB[2 * LDS_BUF];

    const int b = blockIdx.z;
    const int m0 = blockIdx.y * 128, n0 = blockIdx.x * 128;
    const unsigned short* Ab = Ain + (size_t)b * M * Kd;

    const int tid  = threadIdx.x;
    const int lane = tid & 31, w = tid >> 5;
    const int lm = lane & 15, lhi = lane >> 4;
    const int waveM = (w >> 2) * 64;
    const int waveN = (w & 3) * 32;
    const int lrow = tid >> 1, lhalf = tid & 1;

    const unsigned short* ga = Ab + (size_t)(m0 + lrow) * Kd + lhalf * 16;
    const unsigned short* gb = Wt + (size_t)(n0 + lrow) * Kd + lhalf * 16;
    unsigned short* const la = sA + lrow * LDS_STRIDE + lhalf * 16;
    unsigned short* const lb = sB + lrow * LDS_STRIDE + lhalf * 16;

    v8f acc[4][2] = {};
    uint4 pa0, pa1, pb0, pb1;
    {
        const uint4* a4 = reinterpret_cast<const uint4*>(ga);
        const uint4* b4 = reinterpret_cast<const uint4*>(gb);
        pa0 = a4[0]; pa1 = a4[1]; pb0 = b4[0]; pb1 = b4[1];
    }
    reinterpret_cast<uint4*>(la)[0] = pa0; reinterpret_cast<uint4*>(la)[1] = pa1;
    reinterpret_cast<uint4*>(lb)[0] = pb0; reinterpret_cast<uint4*>(lb)[1] = pb1;

    int cur = 0;
    for (int t = 0; t < NT; ++t) {
        __syncthreads();
        if (t + 1 < NT) {
            const uint4* a4 = reinterpret_cast<const uint4*>(ga + (t + 1) * 32);
            const uint4* b4 = reinterpret_cast<const uint4*>(gb + (t + 1) * 32);
            pa0 = a4[0]; pa1 = a4[1]; pb0 = b4[0]; pb1 = b4[1];
        }
        const unsigned short* cA = sA + cur * LDS_BUF;
        const unsigned short* cB = sB + cur * LDS_BUF;
        v16bf bf0 = load_b_frag(cB + (waveN + 0  + lm) * LDS_STRIDE, lhi);
        v16bf bf1 = load_b_frag(cB + (waveN + 16 + lm) * LDS_STRIDE, lhi);
        #pragma unroll
        for (int sm = 0; sm < 4; sm++) {
            v16bf af = load_a_frag(cA + (waveM + sm * 16 + lm) * LDS_STRIDE, lhi);
            acc[sm][0] = __builtin_amdgcn_wmma_f32_16x16x32_bf16(
                false, af, false, bf0, (short)0, acc[sm][0], false, false);
            acc[sm][1] = __builtin_amdgcn_wmma_f32_16x16x32_bf16(
                false, af, false, bf1, (short)0, acc[sm][1], false, false);
        }
        if (t + 1 < NT) {
            unsigned short* na = la + (cur ^ 1) * LDS_BUF;
            unsigned short* nb = lb + (cur ^ 1) * LDS_BUF;
            reinterpret_cast<uint4*>(na)[0] = pa0; reinterpret_cast<uint4*>(na)[1] = pa1;
            reinterpret_cast<uint4*>(nb)[0] = pb0; reinterpret_cast<uint4*>(nb)[1] = pb1;
        }
        cur ^= 1;
    }

    #pragma unroll
    for (int sm = 0; sm < 4; sm++)
        #pragma unroll
        for (int sn = 0; sn < 2; sn++) {
            int gn = n0 + waveN + sn * 16 + lm;
            #pragma unroll
            for (int r = 0; r < 8; r++) {
                int gm = m0 + waveM + sm * 16 + lhi * 8 + r;
                float v = acc[sm][sn][r];
                if (applyGelu) v = 0.5f * v * (1.0f + erff(v * 0.70710678118654752f));
                Dout[((size_t)b * M + gm) * N + gn] = f32_to_bf16_rte(v);
            }
        }
}

// ---------------------------------------------------------------------------
// w[b,q] = sum_o h2[b,q,o] * w3[o]   (wave per row)
// ---------------------------------------------------------------------------
__global__ __launch_bounds__(256)
void matvec_w3_kernel(const unsigned short* __restrict__ H,
                      const float* __restrict__ w3, float* __restrict__ wv,
                      int nrows, int ncols) {
    int wid = threadIdx.x >> 5, lane = threadIdx.x & 31;
    int row = blockIdx.x * 8 + wid;
    if (row >= nrows) return;
    const unsigned short* p = H + (size_t)row * ncols;
    float s = 0.f;
    for (int c = lane; c < ncols; c += 32) s += bf16u_to_f32(p[c]) * w3[c];
    #pragma unroll
    for (int off = 16; off > 0; off >>= 1) s += __shfl_xor(s, off, 32);
    if (lane == 0) wv[row] = s;
}

// ---------------------------------------------------------------------------
// gate[b,k] = tanh( (sum_q wv[b,q] * S[b,q,k]) / 32 )
// ---------------------------------------------------------------------------
__global__ __launch_bounds__(256)
void gate_kernel(const float* __restrict__ wv, const unsigned short* __restrict__ S,
                 float* __restrict__ gate, int LQv, int LKv) {
    int b = blockIdx.y;
    int k = blockIdx.x * 256 + threadIdx.x;
    const unsigned short* Sb = S + (size_t)b * LQv * LKv + k;
    const float* wb = wv + b * LQv;
    float s = 0.f;
    for (int q = 0; q < LQv; q++) s += wb[q] * bf16u_to_f32(Sb[(size_t)q * LKv]);
    gate[b * LKv + k] = tanhf(s * (1.0f / 32.0f));
}

// ---------------------------------------------------------------------------
// out[b,l,c] = value[b,l,c] * (1 + gate[b,l])   (float4 vectorized)
// ---------------------------------------------------------------------------
__global__ __launch_bounds__(256)
void modulate_kernel(const float* __restrict__ V, const float* __restrict__ gate,
                     float* __restrict__ out, size_t total4, int c4) {
    size_t i4 = (size_t)blockIdx.x * 256 + threadIdx.x;
    if (i4 >= total4) return;
    size_t row = i4 / c4;  // b*LK + l
    float g = 1.0f + gate[row];
    float4 v = reinterpret_cast<const float4*>(V)[i4];
    v.x *= g; v.y *= g; v.z *= g; v.w *= g;
    reinterpret_cast<float4*>(out)[i4] = v;
}

// ---------------------------------------------------------------------------
extern "C" void kernel_launch(void* const* d_in, const int* in_sizes, int n_in,
                              void* d_out, int out_size, void* d_ws, size_t ws_size,
                              hipStream_t stream) {
    (void)in_sizes; (void)n_in; (void)out_size; (void)ws_size;
    const int Bv = 16, LQv = 1024, LKv = 1024, Cv = 512;

    const float* query = (const float*)d_in[0];
    const float* key   = (const float*)d_in[1];
    const float* value = (const float*)d_in[2];
    const float* w1    = (const float*)d_in[3];
    const float* w2    = (const float*)d_in[4];
    const float* w3    = (const float*)d_in[5];

    char* ws = (char*)d_ws;
    size_t off = 0;
    auto carve = [&](size_t bytes) -> char* {
        char* p = ws + off;
        off += (bytes + 255) & ~(size_t)255;
        return p;
    };
    unsigned short* S   = (unsigned short*)carve((size_t)Bv * LQv * LKv * 2);
    unsigned short* H1  = (unsigned short*)carve((size_t)Bv * LQv * LKv * 2);
    unsigned short* H2  = (unsigned short*)carve((size_t)Bv * LQv * LKv * 2);
    unsigned short* W1b = (unsigned short*)carve((size_t)LKv * LKv * 2);
    unsigned short* W2b = (unsigned short*)carve((size_t)LKv * LKv * 2);
    float* Qn = (float*)carve((size_t)Bv * LQv * 4);
    float* Kn = (float*)carve((size_t)Bv * LKv * 4);
    float* Wv = (float*)carve((size_t)Bv * LQv * 4);
    float* G  = (float*)carve((size_t)Bv * LKv * 4);

    rownorm_kernel<<<Bv * LQv / 8, 256, 0, stream>>>(query, Qn, Bv * LQv, Cv);
    rownorm_kernel<<<Bv * LKv / 8, 256, 0, stream>>>(key,   Kn, Bv * LKv, Cv);
    cvt_bf16_kernel<<<(LKv * LKv) / 256, 256, 0, stream>>>(w1, W1b, LKv * LKv);
    cvt_bf16_kernel<<<(LKv * LKv) / 256, 256, 0, stream>>>(w2, W2b, LKv * LKv);

    dim3 grid(LKv / 128, LQv / 128, Bv);
    gemm_qk_kernel<<<grid, 256, 0, stream>>>(query, key, Qn, Kn, S);
    gemm_mlp_kernel<<<grid, 256, 0, stream>>>(S,  W1b, H1, 1);
    gemm_mlp_kernel<<<grid, 256, 0, stream>>>(H1, W2b, H2, 1);

    matvec_w3_kernel<<<Bv * LQv / 8, 256, 0, stream>>>(H2, w3, Wv, Bv * LQv, LKv);
    dim3 ggrid(LKv / 256, Bv);
    gate_kernel<<<ggrid, 256, 0, stream>>>(Wv, S, G, LQv, LKv);

    size_t total4 = (size_t)Bv * LKv * Cv / 4;
    modulate_kernel<<<(unsigned)((total4 + 255) / 256), 256, 0, stream>>>(
        value, G, (float*)d_out, total4, Cv / 4);
}
